// SemiFullGN_38302518345746
// MI455X (gfx1250) — compile-verified, matrix operand order checked by reading, not run on hardware
//
#include <hip/hip_runtime.h>
#include <hip/hip_bf16.h>
#include <cstdint>
#include <cstddef>

// ---------------------------------------------------------------------------
// SemiFullGN forward for MI455X (gfx1250, wave32, WMMA).
//
// Heavy math: edge MLP  [E=360000] x (384->128->128->128)  ~59 GFLOP/layer
//             node MLP  [N=30000]  x (256->128->128->128)  ~4  GFLOP/layer
// x3 conv layers -> ~190 GFLOP total. Executed with v_wmma_f32_16x16x32_f16
// (fp16 inputs, fp32 accumulation). Edge activations kept fp16 in HBM
// (2 x 92MB tensors), atom-side activations fp32 for BN fidelity.
// ---------------------------------------------------------------------------

#define N_ATOMS 30000
#define E_PER   12
#define N_EDGES (N_ATOMS * E_PER)      // 360000
#define NCRYS   500
#define APC     (N_ATOMS / NCRYS)      // 60 atoms per crystal
#define ETILES  (N_EDGES / 16)         // 22500
#define NTILES  (N_ATOMS / 16)         // 1875
#define WV      8                      // waves per block in WMMA kernels

typedef __attribute__((ext_vector_type(16))) _Float16 v16h;
typedef __attribute__((ext_vector_type(8)))  float    v8f;

// ---------------------------------------------------------------------------
// WMMA helpers (CDNA5 layouts, CDNA5 ISA §7.12.2)
// A (16x32 f16): lane&15 = M ; kb = (lane&16)?8:0 ; halves 0..7 -> K=kb..kb+7,
//                halves 8..15 -> K=16+kb..16+kb+7
// B (32x16 f16): mirror of A with N = lane&15 (pre-packed by pack_w16)
// C/D (16x16 f32): vgpr g -> M = g + ((lane&16)?8:0), N = lane&15
// ---------------------------------------------------------------------------
__device__ inline v8f wmmaf16(v16h a, v16h b, v8f c) {
  return __builtin_amdgcn_wmma_f32_16x16x32_f16(false, a, false, b, (short)0, c,
                                                false, false);
}

__device__ inline v8f splat8(float v) {
  v8f r;
  r[0] = v; r[1] = v; r[2] = v; r[3] = v;
  r[4] = v; r[5] = v; r[6] = v; r[7] = v;
  return r;
}

// A fragment from fp32 row (p = row_base + k0 + kb), converts 16 floats->halves
__device__ inline v16h afrag_from_f32(const float* p) {
  float4 f0 = *(const float4*)(p);
  float4 f1 = *(const float4*)(p + 4);
  float4 f2 = *(const float4*)(p + 16);
  float4 f3 = *(const float4*)(p + 20);
  v16h a;
  a[0]  = (_Float16)f0.x; a[1]  = (_Float16)f0.y; a[2]  = (_Float16)f0.z; a[3]  = (_Float16)f0.w;
  a[4]  = (_Float16)f1.x; a[5]  = (_Float16)f1.y; a[6]  = (_Float16)f1.z; a[7]  = (_Float16)f1.w;
  a[8]  = (_Float16)f2.x; a[9]  = (_Float16)f2.y; a[10] = (_Float16)f2.z; a[11] = (_Float16)f2.w;
  a[12] = (_Float16)f3.x; a[13] = (_Float16)f3.y; a[14] = (_Float16)f3.z; a[15] = (_Float16)f3.w;
  return a;
}

// A fragment from fp16 row (global or LDS); two 16B loads
__device__ inline v16h afrag_from_f16(const _Float16* p) {
  v16h a;
  *(float4*)&a         = *(const float4*)(p);
  *(((float4*)&a) + 1) = *(const float4*)(p + 16);
  return a;
}

// B fragment from packed weights: frag (kt,nt), lane -> 16 contiguous halves
__device__ inline v16h bfrag_ld(const _Float16* wp, int kt, int nt, int lane) {
  const _Float16* p = wp + (size_t)((((kt << 3) + nt) << 5) + lane) * 16;
  v16h b;
  *(float4*)&b         = *(const float4*)(p);
  *(((float4*)&b) + 1) = *(const float4*)(p + 8);
  return b;
}

// Store one C tile (16x16) into a 16x128 fp16 LDS tile, optional leaky-relu
__device__ inline void store_act_lds(_Float16* buf, int nt, v8f c, int lane,
                                     bool lrelu_on) {
  int col = (nt << 4) + (lane & 15);
  int mb  = (lane & 16) ? 8 : 0;
#pragma unroll
  for (int g = 0; g < 8; ++g) {
    float x = c[g];
    if (lrelu_on) x = (x > 0.f) ? x : 0.2f * x;
    buf[(mb + g) * 128 + col] = (_Float16)x;
  }
}

// One 128->128 layer: A from LDS tile, B from packed weights, bias-initialized C
__device__ inline void wmma_layer_128(const _Float16* in, const _Float16* wp,
                                      const float* bias, int lane, v8f c[8]) {
  int m  = lane & 15;
  int kb = (lane & 16) ? 8 : 0;
#pragma unroll
  for (int nt = 0; nt < 8; ++nt) c[nt] = splat8(bias[(nt << 4) + m]);
#pragma unroll
  for (int kt = 0; kt < 4; ++kt) {
    v16h a = afrag_from_f16(in + m * 128 + kt * 32 + kb);
#pragma unroll
    for (int nt = 0; nt < 8; ++nt) c[nt] = wmmaf16(a, bfrag_ld(wp, kt, nt, lane), c[nt]);
  }
}

// ---------------------------------------------------------------------------
// Weight pre-pack: fp32 [K,128] row-major -> fp16 WMMA B-fragment order
// ---------------------------------------------------------------------------
__global__ void pack_w16(const float* __restrict__ W, int K, _Float16* __restrict__ out) {
  int i    = blockIdx.x * blockDim.x + threadIdx.x; // frag*32 + lane
  int nfrg = (K / 32) * 8;
  if (i >= nfrg * 32) return;
  int lane = i & 31;
  int f    = i >> 5;
  int nt   = f & 7;
  int kt   = f >> 3;
  int kb   = (lane & 16) ? 8 : 0;
  int n    = nt * 16 + (lane & 15);
  _Float16* o = out + (size_t)i * 16;
#pragma unroll
  for (int h = 0; h < 16; ++h) {
    int k = kt * 32 + ((h < 8) ? 0 : 16) + kb + (h & 7);
    o[h]  = (_Float16)W[(size_t)k * 128 + n];
  }
}

// ---------------------------------------------------------------------------
// Embedding kernels (small GEMMs; one row per block, X row staged in LDS)
// ---------------------------------------------------------------------------
__global__ void node_emb_kernel(const float* __restrict__ x, const float* __restrict__ W,
                                const float* __restrict__ b, float* __restrict__ y, int K) {
  __shared__ float xs[128];
  int row = blockIdx.x;
  if ((int)threadIdx.x < K) xs[threadIdx.x] = x[(size_t)row * K + threadIdx.x];
  __syncthreads();
  int col   = threadIdx.x;
  float acc = b[col];
  for (int k = 0; k < K; ++k) acc += xs[k] * W[(size_t)k * 128 + col];
  y[(size_t)row * 128 + col] = acc;
}

__global__ void edge_emb_kernel(const float* __restrict__ x, const float* __restrict__ W,
                                const float* __restrict__ b, _Float16* __restrict__ y, int K) {
  __shared__ float xs[64];
  int row = blockIdx.x;
  if ((int)threadIdx.x < K) xs[threadIdx.x] = x[(size_t)row * K + threadIdx.x];
  __syncthreads();
  int col   = threadIdx.x;
  float acc = b[col];
  for (int k = 0; k < K; ++k) acc += xs[k] * W[(size_t)k * 128 + col];
  y[(size_t)row * 128 + col] = (_Float16)acc;
}

// ---------------------------------------------------------------------------
// Edge MLP: ek = MLP3(phi_e, [atom[e/12] | atom[idx2[e]] | nbr[e]])
// One wave per 16-edge tile; 160 WMMAs/tile.
// ---------------------------------------------------------------------------
__global__ __launch_bounds__(256)
void edge_mlp_wmma(const float* __restrict__ atom, const _Float16* __restrict__ nbr,
                   const int* __restrict__ idx2,
                   const _Float16* __restrict__ w0, const float* __restrict__ b0,
                   const _Float16* __restrict__ w1, const float* __restrict__ b1,
                   const _Float16* __restrict__ w2, const float* __restrict__ b2,
                   _Float16* __restrict__ ek, int ntiles) {
  __shared__ _Float16 sA[WV][16 * 128];
  __shared__ _Float16 sB[WV][16 * 128];
  int wave = threadIdx.x >> 5, lane = threadIdx.x & 31;
  int tile = blockIdx.x * WV + wave;
  if (tile >= ntiles) return;
  int m  = lane & 15;
  int kb = (lane & 16) ? 8 : 0;
  int e  = tile * 16 + m;
  const float*    a1p = atom + (size_t)(e / E_PER) * 128;
  const float*    a2p = atom + (size_t)idx2[e] * 128;
  const _Float16* nbp = nbr + (size_t)e * 128;
  _Float16* A = sA[wave];
  _Float16* B = sB[wave];

  v8f c[8];
  // --- layer 0: K = 384 = [a1(128) | a2(128) | nbr(128)] ---
#pragma unroll
  for (int nt = 0; nt < 8; ++nt) c[nt] = splat8(b0[(nt << 4) + m]);
#pragma unroll
  for (int kt = 0; kt < 12; ++kt) {
    v16h a;
    if (kt < 4)      a = afrag_from_f32(a1p + kt * 32 + kb);
    else if (kt < 8) a = afrag_from_f32(a2p + (kt - 4) * 32 + kb);
    else             a = afrag_from_f16(nbp + (kt - 8) * 32 + kb);
#pragma unroll
    for (int nt = 0; nt < 8; ++nt) c[nt] = wmmaf16(a, bfrag_ld(w0, kt, nt, lane), c[nt]);
  }
#pragma unroll
  for (int nt = 0; nt < 8; ++nt) store_act_lds(A, nt, c[nt], lane, true);

  // --- layer 1 ---
  wmma_layer_128(A, w1, b1, lane, c);
#pragma unroll
  for (int nt = 0; nt < 8; ++nt) store_act_lds(B, nt, c[nt], lane, true);

  // --- layer 2 (no activation) ---
  wmma_layer_128(B, w2, b2, lane, c);
#pragma unroll
  for (int nt = 0; nt < 8; ++nt) store_act_lds(A, nt, c[nt], lane, false);

  // coalesced 16x128 fp16 tile store
  {
    int r = lane >> 1, co = (lane & 1) << 6;
    const _Float16* src = A + r * 128 + co;
    _Float16* dst = ek + ((size_t)tile * 16 + r) * 128 + co;
#pragma unroll
    for (int i = 0; i < 8; ++i) ((float4*)dst)[i] = ((const float4*)src)[i];
  }
}

// ---------------------------------------------------------------------------
// Node MLP: vi = MLP3(phi_v, [atom | rho]) (pre-BN, fp32 out)
// ---------------------------------------------------------------------------
__global__ __launch_bounds__(256)
void node_mlp_wmma(const float* __restrict__ atom, const float* __restrict__ rho,
                   const _Float16* __restrict__ w0, const float* __restrict__ b0,
                   const _Float16* __restrict__ w1, const float* __restrict__ b1,
                   const _Float16* __restrict__ w2, const float* __restrict__ b2,
                   float* __restrict__ vi, int ntiles) {
  __shared__ _Float16 sA[WV][16 * 128];
  __shared__ _Float16 sB[WV][16 * 128];
  int wave = threadIdx.x >> 5, lane = threadIdx.x & 31;
  int tile = blockIdx.x * WV + wave;
  if (tile >= ntiles) return;
  int m  = lane & 15;
  int kb = (lane & 16) ? 8 : 0;
  int row = tile * 16 + m;
  const float* ap = atom + (size_t)row * 128;
  const float* rp = rho + (size_t)row * 128;
  _Float16* A = sA[wave];
  _Float16* B = sB[wave];

  v8f c[8];
  // --- layer 0: K = 256 = [atom(128) | rho(128)] ---
#pragma unroll
  for (int nt = 0; nt < 8; ++nt) c[nt] = splat8(b0[(nt << 4) + m]);
#pragma unroll
  for (int kt = 0; kt < 8; ++kt) {
    v16h a = (kt < 4) ? afrag_from_f32(ap + kt * 32 + kb)
                      : afrag_from_f32(rp + (kt - 4) * 32 + kb);
#pragma unroll
    for (int nt = 0; nt < 8; ++nt) c[nt] = wmmaf16(a, bfrag_ld(w0, kt, nt, lane), c[nt]);
  }
#pragma unroll
  for (int nt = 0; nt < 8; ++nt) store_act_lds(A, nt, c[nt], lane, true);

  wmma_layer_128(A, w1, b1, lane, c);
#pragma unroll
  for (int nt = 0; nt < 8; ++nt) store_act_lds(B, nt, c[nt], lane, true);

  wmma_layer_128(B, w2, b2, lane, c);
  // direct fp32 store (C layout)
  {
    int mb = (lane & 16) ? 8 : 0;
#pragma unroll
    for (int nt = 0; nt < 8; ++nt) {
      int col = (nt << 4) + m;
#pragma unroll
      for (int g = 0; g < 8; ++g)
        vi[(size_t)(tile * 16 + mb + g) * 128 + col] = c[nt][g];
    }
  }
}

// ---------------------------------------------------------------------------
// Edge reduce: rho = segsum(ek)/12 ; nbr += ek (residual, fp16 in place);
//              eksum = segsum(nbr_new)/12.  idx1 = e/12 -> 12 consecutive rows.
// ---------------------------------------------------------------------------
__global__ void edge_reduce_kernel(_Float16* nbr, const _Float16* __restrict__ ek,
                                   float* __restrict__ rho, float* __restrict__ eksum,
                                   int n) {
  int i = blockIdx.x * blockDim.x + threadIdx.x; // atom*128 + col
  if (i >= n) return;
  int atom = i >> 7, col = i & 127;
  size_t base = (size_t)atom * E_PER * 128 + col;
  float rs = 0.f, es = 0.f;
#pragma unroll
  for (int j = 0; j < E_PER; ++j) {
    float ekv = (float)ek[base + (size_t)j * 128];
    float nv  = (float)nbr[base + (size_t)j * 128] + ekv;
    nbr[base + (size_t)j * 128] = (_Float16)nv;
    rs += ekv;
    es += nv;
  }
  rho[i]   = rs * (1.f / E_PER);
  eksum[i] = es * (1.f / E_PER);
}

// ---------------------------------------------------------------------------
// Deterministic BatchNorm stats over 30000 rows x 128 cols (two stage)
// ---------------------------------------------------------------------------
#define BN_CHUNK 256
__global__ void colstats_partial(const float* __restrict__ x, int rows,
                                 float* __restrict__ ps, float* __restrict__ ps2) {
  int col = threadIdx.x;                      // 128 threads
  int r0  = blockIdx.x * BN_CHUNK;
  int r1  = r0 + BN_CHUNK;
  if (r1 > rows) r1 = rows;
  float s = 0.f, s2 = 0.f;
  for (int r = r0; r < r1; ++r) {
    float v = x[(size_t)r * 128 + col];
    s += v; s2 += v * v;
  }
  ps[blockIdx.x * 128 + col]  = s;
  ps2[blockIdx.x * 128 + col] = s2;
}

__global__ void colstats_final(const float* __restrict__ ps, const float* __restrict__ ps2,
                               int nchunks, int rows, float* __restrict__ meanv,
                               float* __restrict__ istdv) {
  int col = threadIdx.x;                      // 128 threads
  float s = 0.f, s2 = 0.f;
  for (int c = 0; c < nchunks; ++c) { s += ps[c * 128 + col]; s2 += ps2[c * 128 + col]; }
  float m   = s / rows;
  float var = s2 / rows - m * m;
  meanv[col] = m;
  istdv[col] = rsqrtf(var + 1e-5f);
}

__global__ void bn_residual_kernel(const float* __restrict__ vi, const float* __restrict__ meanv,
                                   const float* __restrict__ istdv, const float* __restrict__ g,
                                   const float* __restrict__ b, float* atom, int n) {
  int i = blockIdx.x * blockDim.x + threadIdx.x;
  if (i >= n) return;
  int col = i & 127;
  atom[i] = atom[i] + g[col] * (vi[i] - meanv[col]) * istdv[col] + b[col];
}

// ---------------------------------------------------------------------------
// Crystal pooling: gf[c, :] = (1/60) * sum over 60 atoms of [vi_new | eksum]
// ---------------------------------------------------------------------------
__global__ void crys_sum_kernel(const float* __restrict__ vi, const float* __restrict__ eksum,
                                float* __restrict__ gf) {
  int i = blockIdx.x * blockDim.x + threadIdx.x; // crys*256 + col
  if (i >= NCRYS * 256) return;
  int c = i >> 8, col = i & 255;
  const float* src = (col < 128) ? (vi + (size_t)c * APC * 128 + col)
                                 : (eksum + (size_t)c * APC * 128 + (col - 128));
  float s = 0.f;
  for (int a = 0; a < APC; ++a) s += src[(size_t)a * 128];
  gf[i] = s * (1.f / APC);
}

// ---------------------------------------------------------------------------
// Tail (500-row MLPs) -- tiny; plain kernels
// ---------------------------------------------------------------------------
#define ACT_NONE  0
#define ACT_LRELU 1
#define ACT_TANH  2
__global__ void linear_small(const float* __restrict__ X, const float* __restrict__ W,
                             const float* __restrict__ b, float* __restrict__ Y,
                             int M, int K, int Nc, int act) {
  int i = blockIdx.x * blockDim.x + threadIdx.x;
  if (i >= M * Nc) return;
  int m = i / Nc, n = i - m * Nc;
  const float* xr = X + (size_t)m * K;
  float acc = b ? b[n] : 0.f;
  for (int k = 0; k < K; ++k) acc += xr[k] * W[(size_t)k * Nc + n];
  if (act == ACT_LRELU) acc = (acc > 0.f) ? acc : 0.2f * acc;
  else if (act == ACT_TANH) acc = tanhf(acc);
  Y[i] = acc;
}

__global__ void bn_tail_kernel(float* x, const float* __restrict__ g,
                               const float* __restrict__ b, int rows, int cols) {
  int col = threadIdx.x;
  if (col >= cols) return;
  float s = 0.f, s2 = 0.f;
  for (int r = 0; r < rows; ++r) {
    float v = x[(size_t)r * cols + col];
    s += v; s2 += v * v;
  }
  float m   = s / rows;
  float var = s2 / rows - m * m;
  float is  = rsqrtf(var + 1e-5f);
  for (int r = 0; r < rows; ++r)
    x[(size_t)r * cols + col] = g[col] * (x[(size_t)r * cols + col] - m) * is + b[col];
}

__global__ void concat_zz_kernel(const float* __restrict__ z, const float* __restrict__ ur,
                                 const float* __restrict__ rr, float* __restrict__ zz) {
  int i = blockIdx.x * blockDim.x + threadIdx.x; // 500*512
  if (i >= NCRYS * 512) return;
  int c = i >> 9, col = i & 511;
  float v;
  if (col < 256)      v = z[c * 256 + col];
  else if (col < 384) v = ur[c * 128 + (col - 256)];
  else                v = rr[c * 128 + (col - 384)];
  zz[i] = v;
}

__global__ void cell_mm_kernel(const float* __restrict__ f, const float* __restrict__ cell,
                               float* __restrict__ o) {
  int i = blockIdx.x * blockDim.x + threadIdx.x; // 500*192
  if (i >= NCRYS * 192) return;
  int c = i / 192, rj = i - c * 192;
  int r = rj / 3, j = rj - r * 3;
  const float* fr = f + (size_t)c * 192 + r * 3;
  const float* cl = cell + (size_t)c * 9;
  o[i] = fr[0] * cl[j] + fr[1] * cl[3 + j] + fr[2] * cl[6 + j];
}

// ---------------------------------------------------------------------------
// Host side
// ---------------------------------------------------------------------------
static inline void* wscur(char*& cur, size_t bytes) {
  void* p = cur;
  cur += (bytes + 255) & ~(size_t)255;
  return p;
}

// Input index table: params flattened JAX-style (sorted dict keys, lists in
// order), then remaining setup_inputs entries in dict order.
//  0 conv_to_fc.W  1 conv_to_fc.b
//  2+14*l : convs[l] = {bn_b, bn_g, eW0,eb0,eW1,eb1,eW2,eb2, vW0,vb0,vW1,vb1,vW2,vb2}
// 44 delta_emb.W  45 edge_emb.W  46 edge_emb.b
// 47..52 fc_out[0..2].{W,b}  53..56 fcs[0..1].{W,b}
// 57 node_emb.W  58 node_emb.b  59..62 out_layer[0..1].{W,b}
// 63..66 phi_u[0..1].{W,b}  67 zcat.bn_b  68 zcat.bn_g  69 zcat.lin.W  70 zcat.lin.b
// 71 atom_fea  72 nbr_fea  73 idx1  74 idx2  75 num_nbrs  76 crys_idx
// 77 unrelaxed  78 relaxed  79 cell  80 delta
extern "C" void kernel_launch(void* const* d_in, const int* in_sizes, int n_in,
                              void* d_out, int out_size, void* d_ws, size_t ws_size,
                              hipStream_t stream) {
  (void)in_sizes; (void)n_in; (void)out_size; (void)ws_size;
  auto F = [&](int i) { return (const float*)d_in[i]; };
  const int* idx2 = (const int*)d_in[74];

  char* cur = (char*)d_ws;
  _Float16* nbr_h   = (_Float16*)wscur(cur, (size_t)N_EDGES * 128 * 2);
  _Float16* ek_h    = (_Float16*)wscur(cur, (size_t)N_EDGES * 128 * 2);
  float*    atom_f  = (float*)wscur(cur, (size_t)N_ATOMS * 128 * 4);
  float*    rho_f   = (float*)wscur(cur, (size_t)N_ATOMS * 128 * 4);
  float*    vi_f    = (float*)wscur(cur, (size_t)N_ATOMS * 128 * 4);
  float*    eksum_f = (float*)wscur(cur, (size_t)N_ATOMS * 128 * 4);
  float*    gf      = (float*)wscur(cur, (size_t)NCRYS * 256 * 4);
  float*    z_buf   = (float*)wscur(cur, (size_t)NCRYS * 256 * 4);
  float*    ta      = (float*)wscur(cur, (size_t)NCRYS * 512 * 4);
  float*    tb      = (float*)wscur(cur, (size_t)NCRYS * 512 * 4);
  float*    psum    = (float*)wscur(cur, (size_t)128 * 128 * 4);
  float*    psumsq  = (float*)wscur(cur, (size_t)128 * 128 * 4);
  float*    meanb   = (float*)wscur(cur, 128 * 4);
  float*    istdb   = (float*)wscur(cur, 128 * 4);
  // packed fp16 conv weights: per layer {eW0(384),eW1(128),eW2(128),vW0(256),vW1(128),vW2(128)}
  static const int wk[6] = {384, 128, 128, 256, 128, 128};
  _Float16* wpk[3][6];
  for (int l = 0; l < 3; ++l)
    for (int j = 0; j < 6; ++j)
      wpk[l][j] = (_Float16*)wscur(cur, (size_t)wk[j] * 128 * 2);

  // --- pack all conv weights to WMMA B-fragment order ---
  static const int woff[6] = {2, 4, 6, 8, 10, 12}; // offsets within conv block
  for (int l = 0; l < 3; ++l) {
    for (int j = 0; j < 6; ++j) {
      int nth = (wk[j] / 32) * 8 * 32;
      pack_w16<<<(nth + 255) / 256, 256, 0, stream>>>(F(2 + 14 * l + woff[j]), wk[j], wpk[l][j]);
    }
  }

  // --- embeddings ---
  node_emb_kernel<<<N_ATOMS, 128, 0, stream>>>(F(71), F(57), F(58), atom_f, 92);
  edge_emb_kernel<<<N_EDGES, 128, 0, stream>>>(F(72), F(45), F(46), nbr_h, 41);

  // --- 3 conv layers ---
  const int nchunks = (N_ATOMS + BN_CHUNK - 1) / BN_CHUNK;
  for (int l = 0; l < 3; ++l) {
    int pb = 2 + 14 * l;
    const float* bnb = F(pb + 0);
    const float* bng = F(pb + 1);
    edge_mlp_wmma<<<(ETILES + WV - 1) / WV, 32 * WV, 0, stream>>>(
        atom_f, nbr_h, idx2, wpk[l][0], F(pb + 3), wpk[l][1], F(pb + 5),
        wpk[l][2], F(pb + 7), ek_h, ETILES);
    edge_reduce_kernel<<<(N_ATOMS * 128 + 255) / 256, 256, 0, stream>>>(
        nbr_h, ek_h, rho_f, eksum_f, N_ATOMS * 128);
    node_mlp_wmma<<<(NTILES + WV - 1) / WV, 32 * WV, 0, stream>>>(
        atom_f, rho_f, wpk[l][3], F(pb + 9), wpk[l][4], F(pb + 11),
        wpk[l][5], F(pb + 13), vi_f, NTILES);
    colstats_partial<<<nchunks, 128, 0, stream>>>(vi_f, N_ATOMS, psum, psumsq);
    colstats_final<<<1, 128, 0, stream>>>(psum, psumsq, nchunks, N_ATOMS, meanb, istdb);
    bn_residual_kernel<<<(N_ATOMS * 128 + 255) / 256, 256, 0, stream>>>(
        vi_f, meanb, istdb, bng, bnb, atom_f, N_ATOMS * 128);
    if (l == 2)
      crys_sum_kernel<<<(NCRYS * 256 + 255) / 256, 256, 0, stream>>>(atom_f, eksum_f, gf);
  }

  // --- tail MLPs on [500, *] ---
  float* outp = (float*)d_out;
  auto lin = [&](const float* X, int pw, int pbias, float* Y, int M, int K, int Nc, int act) {
    linear_small<<<(M * Nc + 255) / 256, 256, 0, stream>>>(X, F(pw),
        pbias >= 0 ? F(pbias) : nullptr, Y, M, K, Nc, act);
  };
  lin(gf, 63, 64, ta, NCRYS, 256, 256, ACT_LRELU);                 // phi_u[0]
  lin(ta, 65, 66, z_buf, NCRYS, 256, 256, ACT_TANH);               // phi_u[1] -> z
  concat_zz_kernel<<<(NCRYS * 512 + 255) / 256, 256, 0, stream>>>(z_buf, F(77), F(78), tb);
  lin(tb, 69, 70, ta, NCRYS, 512, 256, ACT_LRELU);                 // z_cat.lin
  bn_tail_kernel<<<1, 256, 0, stream>>>(ta, F(68), F(67), NCRYS, 256); // z_cat bn (g,b)
  lin(ta, 0, 1, tb, NCRYS, 256, 256, ACT_LRELU);                   // conv_to_fc
  lin(tb, 53, 54, ta, NCRYS, 256, 256, ACT_LRELU);                 // fcs[0]
  lin(ta, 55, 56, tb, NCRYS, 256, 256, ACT_LRELU);                 // fcs[1]
  lin(tb, 47, 48, ta, NCRYS, 256, 256, ACT_LRELU);                 // fc_out[0]
  lin(ta, 49, 50, tb, NCRYS, 256, 256, ACT_LRELU);                 // fc_out[1]
  lin(tb, 51, 52, ta, NCRYS, 256, 192, ACT_LRELU);                 // fc_out[2]
  cell_mm_kernel<<<(NCRYS * 192 + 255) / 256, 256, 0, stream>>>(ta, F(79), tb);
  lin(tb, 59, 60, ta, NCRYS, 192, 128, ACT_LRELU);                 // out_layer[0]
  lin(ta, 61, 62, outp, NCRYS, 128, 9, ACT_NONE);                  // out_layer[1] -> out
  hipMemcpyAsync(outp + NCRYS * 9, z_buf, (size_t)NCRYS * 256 * sizeof(float),
                 hipMemcpyDeviceToDevice, stream);                 // second output: z
}
